// QuantLinear_75230647157288
// MI455X (gfx1250) — matrix operand order repaired and checked by reading, not compiled, not run on hardware
//
#include <hip/hip_runtime.h>
#include <hip/hip_bf16.h>

#define IN_FEATURES  8192
#define OUT_FEATURES 8192
#define NROWS        128      // 32*4 batch rows
#define TN           64       // N tile per block
#define TK           64       // K chunk
#define LDSPAD       72       // padded row stride in halves (breaks bank-conflict strides)

typedef __attribute__((ext_vector_type(16))) _Float16 v16h;
typedef __attribute__((ext_vector_type(8)))  float    v8f;

union U16H { v16h v; uint4 q[2]; };

// ---------------------------------------------------------------------------
// Kernel 1: x = fwht(x / scaleWH * SU)   (fp16 in -> fp16 out, fp32 LDS math)
// ---------------------------------------------------------------------------
__global__ __launch_bounds__(256) void fwht_pre_kernel(
    const _Float16* __restrict__ in,
    const _Float16* __restrict__ scaleWH,
    const _Float16* __restrict__ SU,
    _Float16* __restrict__ xh)
{
    __shared__ float s[IN_FEATURES];
    const int r = blockIdx.x;
    const int t = threadIdx.x;
    const _Float16* row = in + (size_t)r * IN_FEATURES;

    #pragma unroll
    for (int i = 0; i < IN_FEATURES / 256; ++i) {
        int idx = t + i * 256;
        float v = (float)row[idx] / (float)scaleWH[idx] * (float)SU[idx];
        s[idx] = v;
    }
    __syncthreads();

    for (int h = 1; h < IN_FEATURES; h <<= 1) {
        #pragma unroll
        for (int p = 0; p < IN_FEATURES / 512; ++p) {
            int pair = t + p * 256;
            int i0 = ((pair & ~(h - 1)) << 1) | (pair & (h - 1));
            float a = s[i0], b = s[i0 + h];
            s[i0]     = a + b;
            s[i0 + h] = a - b;
        }
        __syncthreads();
    }

    const float sc = 0.011048543456039804f;               // 1/sqrt(8192)
    _Float16* orow = xh + (size_t)r * IN_FEATURES;
    #pragma unroll
    for (int i = 0; i < IN_FEATURES / 256; ++i) {
        int idx = t + i * 256;
        orow[idx] = (_Float16)(s[idx] * sc);
    }
}

// ---------------------------------------------------------------------------
// Async-copy one thread's slice of the A tile (4 x b128) into LDS buffer.
// Uses GLOBAL_LOAD_ASYNC_TO_LDS_B128 (ASYNCcnt path, bypasses VGPRs).
// LDS byte offset = low 32 bits of the generic (flat) LDS address.
// ---------------------------------------------------------------------------
__device__ __forceinline__ void issue_async_a_tile(
    const _Float16* __restrict__ xh, _Float16* sAbuf, int k0, int t)
{
    unsigned base = (unsigned)(uintptr_t)sAbuf;
    #pragma unroll
    for (int i = 0; i < 4; ++i) {
        int lin = (i * 256 + t) * 8;            // half index within 128x64 tile
        int m = lin >> 6;
        int c = lin & 63;
        unsigned ldsoff = base + (unsigned)((m * LDSPAD + c) * 2);
        unsigned long long ga =
            (unsigned long long)(uintptr_t)(xh + (size_t)m * IN_FEATURES + k0 + c);
        asm volatile("global_load_async_to_lds_b128 %0, %1, off"
                     :: "v"(ldsoff), "v"(ga) : "memory");
    }
}

// ---------------------------------------------------------------------------
// Kernel 2: out_f32 = x_fp16 @ W_hat^T with on-the-fly codebook decompression.
// Double-buffered LDS, async A staging, software-pipelined code fetch.
// grid.x = OUT_FEATURES/TN = 128 blocks, 256 threads = 8 waves.
// ---------------------------------------------------------------------------
__global__ __launch_bounds__(256) void qgemm_kernel(
    const _Float16* __restrict__ xh,        // [128][8192] fp16 (post-FWHT)
    const int*      __restrict__ qidx,      // [8192][1024] codes 0..255
    const _Float16* __restrict__ cb,        // [256][8] codebook
    const _Float16* __restrict__ wscale,    // scalar
    float*          __restrict__ outf)      // [128][8192] f32
{
    __shared__ __align__(16) _Float16 sCB[256 * 8];           // scaled codebook
    __shared__ __align__(16) _Float16 sA[2][NROWS * LDSPAD];  // A tiles (dbl buf)
    __shared__ __align__(16) _Float16 sW[2][TN * LDSPAD];     // W tiles (dbl buf)

    const int t    = threadIdx.x;
    const int lane = t & 31;
    const int w    = t >> 5;
    const int n0   = blockIdx.x * TN;
    constexpr int NCHUNK = IN_FEATURES / TK;

    // fold Wscale into the codebook once
    {
        const float ws = (float)wscale[0];
        const _Float16* src = cb + t * 8;
        #pragma unroll
        for (int j = 0; j < 8; ++j)
            sCB[t * 8 + j] = (_Float16)((float)src[j] * ws);
    }

    v8f acc[4];
    {
        v8f z = {0.f, 0.f, 0.f, 0.f, 0.f, 0.f, 0.f, 0.f};
        #pragma unroll
        for (int i = 0; i < 4; ++i) acc[i] = z;
    }

    const int m0    = w * 16;
    const int mrow  = m0 + (lane & 15);
    const int krow  = (lane < 16) ? 0 : 8;    // A-fragment K sub-offset (ISA layout)
    const int koffB = (lane < 16) ? 0 : 8;    // B-fragment K sub-offset (ISA layout)

    // ---- prologue: async A(0); fetch codes(0) into registers
    issue_async_a_tile(xh, &sA[0][0], 0, t);

    const int  nrow  = t >> 1;
    const int  cbase = (t & 1) * 4;
    const int* qbase = qidx + (size_t)(n0 + nrow) * (IN_FEATURES / 8) + cbase;
    int4 codes_cur = {0, 0, 0, 0};
    if (t < 128) codes_cur = *(const int4*)qbase;

    __syncthreads();   // sCB visible to all

    int b = 0;
    for (int kt = 0; kt < NCHUNK; ++kt) {
        // own async A(kt) slice resident before signaling the barrier below
        asm volatile("s_wait_asynccnt 0x0" ::: "memory");

        // decompress W(kt) from pre-fetched codes into sW[b]
        if (t < 128) {
            int cc[4] = {codes_cur.x, codes_cur.y, codes_cur.z, codes_cur.w};
            #pragma unroll
            for (int j = 0; j < 4; ++j) {
                int idx = cc[j] & 255;
                *(uint4*)&sW[b][nrow * LDSPAD + (cbase + j) * 8] =
                    *(const uint4*)&sCB[idx * 8];
            }
        }
        // pipeline: fetch codes(kt+1) while computing chunk kt
        if (kt + 1 < NCHUNK && t < 128)
            codes_cur = *(const int4*)(qbase + (size_t)(kt + 1) * (TK / 8));
        // warm L2 two chunks ahead
        if (kt + 2 < NCHUNK && t < TN)
            __builtin_prefetch(qidx + (size_t)(n0 + t) * (IN_FEATURES / 8)
                               + (size_t)(kt + 2) * (TK / 8), 0, 0);

        __syncthreads();   // whole A(kt) tile + W(kt) tile visible

        // kick off next A tile into the other buffer while we compute
        if (kt + 1 < NCHUNK)
            issue_async_a_tile(xh, &sA[b ^ 1][0], (kt + 1) * TK, t);

        // ---- compute: 2 K-steps x 4 N-subtiles = 8 WMMAs per wave per chunk
        #pragma unroll
        for (int kk = 0; kk < TK; kk += 32) {
            U16H a;
            a.q[0] = *(const uint4*)&sA[b][mrow * LDSPAD + kk + krow];
            a.q[1] = *(const uint4*)&sA[b][mrow * LDSPAD + kk + 16 + krow];
            #pragma unroll
            for (int ns = 0; ns < 4; ++ns) {
                int n = ns * 16 + (lane & 15);
                U16H bb;
                bb.q[0] = *(const uint4*)&sW[b][n * LDSPAD + kk + koffB];
                bb.q[1] = *(const uint4*)&sW[b][n * LDSPAD + kk + koffB + 16];
                acc[ns] = __builtin_amdgcn_wmma_f32_16x16x32_f16(
                    false, a.v, false, bb.v, (short)0, acc[ns], false, false);
            }
        }
        b ^= 1;
    }

    // ---- store accumulators (C/D layout: lane N=lane&15, VGPR r -> M r or r+8)
    #pragma unroll
    for (int ns = 0; ns < 4; ++ns) {
        int col = n0 + ns * 16 + (lane & 15);
        #pragma unroll
        for (int r = 0; r < 8; ++r) {
            int row = m0 + r + ((lane < 16) ? 0 : 8);
            outf[(size_t)row * OUT_FEATURES + col] = acc[ns][r];
        }
    }
}

// ---------------------------------------------------------------------------
// Kernel 3: out = fwht(gemm_out) * SV + bias   (f32 in -> fp16 out)
// ---------------------------------------------------------------------------
__global__ __launch_bounds__(256) void fwht_post_kernel(
    const float*    __restrict__ inf,
    const _Float16* __restrict__ SV,
    const _Float16* __restrict__ bias,
    _Float16* __restrict__ out)
{
    __shared__ float s[OUT_FEATURES];
    const int r = blockIdx.x;
    const int t = threadIdx.x;
    const float* row = inf + (size_t)r * OUT_FEATURES;

    #pragma unroll
    for (int i = 0; i < OUT_FEATURES / 256; ++i) {
        int idx = t + i * 256;
        s[idx] = row[idx];
    }
    __syncthreads();

    for (int h = 1; h < OUT_FEATURES; h <<= 1) {
        #pragma unroll
        for (int p = 0; p < OUT_FEATURES / 512; ++p) {
            int pair = t + p * 256;
            int i0 = ((pair & ~(h - 1)) << 1) | (pair & (h - 1));
            float a = s[i0], b = s[i0 + h];
            s[i0]     = a + b;
            s[i0 + h] = a - b;
        }
        __syncthreads();
    }

    const float sc = 0.011048543456039804f;               // 1/sqrt(8192)
    _Float16* orow = out + (size_t)r * OUT_FEATURES;
    #pragma unroll
    for (int i = 0; i < OUT_FEATURES / 256; ++i) {
        int idx = t + i * 256;
        orow[idx] = (_Float16)(s[idx] * sc * (float)SV[idx] + (float)bias[idx]);
    }
}

// ---------------------------------------------------------------------------
extern "C" void kernel_launch(void* const* d_in, const int* in_sizes, int n_in,
                              void* d_out, int out_size, void* d_ws, size_t ws_size,
                              hipStream_t stream) {
    const _Float16* input   = (const _Float16*)d_in[0];
    const int*      qidx    = (const int*)d_in[1];
    const _Float16* cb      = (const _Float16*)d_in[2];
    const _Float16* scaleWH = (const _Float16*)d_in[3];
    const _Float16* SU      = (const _Float16*)d_in[4];
    const _Float16* SV      = (const _Float16*)d_in[5];
    const _Float16* wscale  = (const _Float16*)d_in[6];
    const _Float16* bias    = (const _Float16*)d_in[7];
    _Float16* out = (_Float16*)d_out;

    // workspace: fp16 x (2 MB) | f32 gemm out (4 MB)
    _Float16* xh   = (_Float16*)d_ws;
    float*    outf = (float*)((char*)d_ws + (size_t)NROWS * IN_FEATURES * sizeof(_Float16));

    fwht_pre_kernel<<<NROWS, 256, 0, stream>>>(input, scaleWH, SU, xh);
    qgemm_kernel<<<OUT_FEATURES / TN, 256, 0, stream>>>(xh, qidx, cb, wscale, outf);
    fwht_post_kernel<<<NROWS, 256, 0, stream>>>(outf, SV, bias, out);
}
